// STGAT_46351287059145
// MI455X (gfx1250) — compile-verified, matrix operand order checked by reading, not run on hardware
//
#include <hip/hip_runtime.h>
#include <hip/hip_bf16.h>

// ---------------------------------------------------------------------------
// Types for CDNA5 WMMA (wave32, 16x16x32 bf16 -> f32)
// ---------------------------------------------------------------------------
typedef __attribute__((ext_vector_type(16))) __bf16       bf16x16;
typedef __attribute__((ext_vector_type(8)))  float        f32x8;
typedef __attribute__((ext_vector_type(4)))  float        f32x4;
typedef __attribute__((ext_vector_type(4)))  unsigned int u32x4;

// Force global (addrspace 1) vector loads: operands all live in device
// global memory; the generic->global cast via integer keeps the backend on
// global_load_b128 (LOADcnt only) instead of flat_load_b128 (LOADcnt+DScnt).
// Plain ext_vector primitives (not HIP_vector_type classes) so the AS-
// qualified dereference needs no constructor.
#define GAS __attribute__((address_space(1)))

__device__ __forceinline__ u32x4 load_g_u4(const void* p)
{
    return *(const GAS u32x4*)(unsigned long long)p;
}
__device__ __forceinline__ f32x4 load_g_f4(const float* p)
{
    return *(const GAS f32x4*)(unsigned long long)p;
}

__device__ __forceinline__ void load8f(const float* __restrict__ p, float* d)
{
    const f32x4 a = load_g_f4(p);
    const f32x4 b = load_g_f4(p + 4);
    #pragma unroll
    for (int e = 0; e < 4; ++e) { d[e] = a[e]; d[4 + e] = b[e]; }
}

// B operands are pre-swizzled into WMMA-fragment order:
//   Bp[((k >> 5) * N + col) * 32 + (k & 31)]
// so a lane's 16 fragment elements are two contiguous 16B chunks
// (offsets hi8 and 16+hi8 within the 32-element K-group).
__device__ __forceinline__ bf16x16 load_frag16(const __bf16* __restrict__ p)
{
    union { u32x4 u; __bf16 h[8]; } c0, c1;
    c0.u = load_g_u4(p);
    c1.u = load_g_u4(p + 16);
    bf16x16 f;
    #pragma unroll
    for (int e = 0; e < 8; ++e) { f[e] = c0.h[e]; f[8 + e] = c1.h[e]; }
    return f;
}

// ---------------------------------------------------------------------------
// Generic GEMM: C[M,N] = A[M,K] (bf16 row-major, optional row shift within
// segments of length Lseg for causal convs) * Bp (fragment-packed bf16).
// Block = 8 waves, 256x64 tile; each wave owns 32 rows (2 strips, 8 WMMAs /
// K-step). Invalid rows/cols read a zero-filled scratch buffer with stride 0,
// so the K-loop is fully branch-free (pure pointer induction, no EXEC
// juggling, no uniform branches). beta=1 accumulates, bias added when
// non-null. Requires K % 32 == 0.
// ---------------------------------------------------------------------------
__global__ void __launch_bounds__(256)
gemm_bf16_wmma(const __bf16* __restrict__ A, const __bf16* __restrict__ Bp,
               const __bf16* __restrict__ zbuf,
               float* __restrict__ C, const float* __restrict__ bias,
               int M, int N, int K, int Lseg, int shift, int beta)
{
    const int tid  = threadIdx.x;
    const int wave = tid >> 5;
    const int lane = tid & 31;
    const int lr   = lane & 15;
    const int hi8  = (lane >> 4) << 3;      // 0 or 8

    const int rowTile = blockIdx.x * 256;
    const int colTile = blockIdx.y * 64;

    // Loop-invariant operand pointers; invalid lanes -> zbuf with step 0.
    const __bf16* aPtr[2];
    int aStep[2];
    #pragma unroll
    for (int s = 0; s < 2; ++s) {
        const int row  = rowTile + wave * 32 + s * 16 + lr;
        const int bseg = row / Lseg;
        const int l    = row - bseg * Lseg;
        const int ls   = l + shift;
        const bool rv  = (row < M) && (ls >= 0) && (ls < Lseg);
        aPtr[s]  = rv ? (A + (size_t)(bseg * Lseg + ls) * K + hi8) : zbuf;
        aStep[s] = rv ? 32 : 0;
    }
    const __bf16* bPtr[4];
    int bStep[4];
    #pragma unroll
    for (int nt = 0; nt < 4; ++nt) {
        const int col = colTile + nt * 16 + lr;
        const bool cv = (col < N);
        bPtr[nt]  = cv ? (Bp + ((size_t)col << 5) + hi8) : zbuf;
        bStep[nt] = cv ? (N << 5) : 0;
    }

    f32x8 acc[2][4];
    {
        f32x8 z = {};
        #pragma unroll
        for (int s = 0; s < 2; ++s)
            #pragma unroll
            for (int nt = 0; nt < 4; ++nt) acc[s][nt] = z;
    }

    for (int kk = 0; kk < K; kk += 32) {
        bf16x16 af[2];
        #pragma unroll
        for (int s = 0; s < 2; ++s) {
            af[s] = load_frag16(aPtr[s]);
            __builtin_prefetch(aPtr[s] + 64, 0, 1);
            aPtr[s] += aStep[s];
        }
        #pragma unroll
        for (int nt = 0; nt < 4; ++nt) {
            const bf16x16 bf = load_frag16(bPtr[nt]);
            bPtr[nt] += bStep[nt];
            #pragma unroll
            for (int s = 0; s < 2; ++s)
                acc[s][nt] = __builtin_amdgcn_wmma_f32_16x16x32_bf16(
                    false, af[s], false, bf, (short)0, acc[s][nt], false, false);
        }
    }

    // C/D layout: lanes 0-15 -> M = v, lanes 16-31 -> M = 8 + v
    #pragma unroll
    for (int s = 0; s < 2; ++s) {
        const int rbase = rowTile + wave * 32 + s * 16 + hi8;
        #pragma unroll
        for (int nt = 0; nt < 4; ++nt) {
            const int col = colTile + nt * 16 + lr;
            if (col >= N) continue;
            #pragma unroll
            for (int v = 0; v < 8; ++v) {
                const int r = rbase + v;
                if (r >= M) continue;
                float val = acc[s][nt][v];
                if (bias) val += bias[col];
                const size_t o = (size_t)r * N + col;
                if (beta) C[o] += val; else C[o] = val;
            }
        }
    }
}

// ---------------------------------------------------------------------------
// Fused attention GEMM: Hp[b,i,:] = sum_j p[i,j] * H[b,j,:]
// p[i,j] = exp(adj[i,j]*lrelu(f1[i]+f2[j]) - m[j]) / s[j] built in registers
// (A-fragment). H is fragment-packed. No (N,N) matrix, no LDS. y-grid = batch.
// ---------------------------------------------------------------------------
__global__ void __launch_bounds__(256)
att_gemm_wmma(const float* __restrict__ adj,
              const float* __restrict__ f1, const float* __restrict__ f2,
              const float* __restrict__ cmax, const float* __restrict__ rsum,
              const __bf16* __restrict__ Hpk, float* __restrict__ Hp, int Nn)
{
    const int b = blockIdx.y;
    const float*  f1b = f1   + b * Nn;
    const float*  f2b = f2   + b * Nn;
    const float*  cmb = cmax + b * Nn;
    const float*  rsb = rsum + b * Nn;
    const __bf16* Hkb = Hpk + (size_t)b * Nn * 64;   // Nn multiple of 32
    float*        Hpb = Hp  + (size_t)b * Nn * 64;

    const int tid  = threadIdx.x;
    const int wave = tid >> 5;
    const int lane = tid & 31;
    const int lr   = lane & 15;
    const int hi8  = (lane >> 4) << 3;

    const int i = blockIdx.x * 128 + wave * 16 + lr;
    const float f1i = f1b[i];
    const float* adji = adj + (size_t)i * Nn;

    // Loop-invariant packed-H column pointers (step = 64 cols * 32 = 2048).
    const __bf16* hPtr[4];
    #pragma unroll
    for (int nt = 0; nt < 4; ++nt)
        hPtr[nt] = Hkb + (((size_t)(nt * 16 + lr)) << 5) + hi8;

    f32x8 acc[4];
    {
        f32x8 z = {};
        #pragma unroll
        for (int nt = 0; nt < 4; ++nt) acc[nt] = z;
    }

    for (int jj = 0; jj < Nn; jj += 32) {
        float va[16], vf2[16], vm[16], vrs[16];
        load8f(adji + jj + hi8,      va);
        load8f(adji + jj + 16 + hi8, va + 8);
        load8f(f2b  + jj + hi8,      vf2);
        load8f(f2b  + jj + 16 + hi8, vf2 + 8);
        load8f(cmb  + jj + hi8,      vm);
        load8f(cmb  + jj + 16 + hi8, vm + 8);
        load8f(rsb  + jj + hi8,      vrs);
        load8f(rsb  + jj + 16 + hi8, vrs + 8);
        if (jj + 32 < Nn) __builtin_prefetch(adji + jj + 64, 0, 1);

        bf16x16 af;
        #pragma unroll
        for (int e = 0; e < 16; ++e) {
            const float t  = f1i + vf2[e];
            const float zz = va[e] * (t > 0.0f ? t : 0.2f * t);
            af[e] = (__bf16)(__expf(zz - vm[e]) * vrs[e]);
        }

        #pragma unroll
        for (int nt = 0; nt < 4; ++nt) {
            const bf16x16 bf = load_frag16(hPtr[nt]);
            hPtr[nt] += 64 << 5;
            acc[nt] = __builtin_amdgcn_wmma_f32_16x16x32_bf16(
                false, af, false, bf, (short)0, acc[nt], false, false);
        }
    }

    const int rbase = blockIdx.x * 128 + wave * 16 + hi8;
    #pragma unroll
    for (int nt = 0; nt < 4; ++nt) {
        const int col = nt * 16 + lr;
        #pragma unroll
        for (int v = 0; v < 8; ++v)
            Hpb[(size_t)(rbase + v) * 64 + col] = acc[nt][v];
    }
}

// ---------------------------------------------------------------------------
// Per-column online softmax statistics over axis i (softmax over dim=1).
// Threads own consecutive columns j -> coalesced adj row reads each step.
// ---------------------------------------------------------------------------
__global__ void col_stats(const float* __restrict__ adj, const float* __restrict__ f1,
                          const float* __restrict__ f2, float* __restrict__ cmax,
                          float* __restrict__ rsum, int Nn)
{
    const int b = blockIdx.y;
    const int j = blockIdx.x * blockDim.x + threadIdx.x;
    const float* f1b = f1 + b * Nn;
    const float  f2j = f2[b * Nn + j];
    float m = -3.0e38f, s = 0.0f;
    for (int i = 0; i < Nn; ++i) {
        const float t  = f1b[i] + f2j;
        const float zz = adj[(size_t)i * Nn + j] * (t > 0.0f ? t : 0.2f * t);
        const float mn = fmaxf(m, zz);
        s = s * __expf(m - mn) + __expf(zz - mn);
        m = mn;
    }
    cmax[b * Nn + j] = m;
    rsum[b * Nn + j] = 1.0f / s;
}

// ---------------------------------------------------------------------------
// f1 = h @ a1, f2 = h @ a2 (per row of h)
// ---------------------------------------------------------------------------
__global__ void f12_kernel(const float* __restrict__ H, const float* __restrict__ a1,
                           const float* __restrict__ a2, float* __restrict__ f1,
                           float* __restrict__ f2, int total)
{
    const int n = blockIdx.x * blockDim.x + threadIdx.x;
    if (n >= total) return;
    const float* hr = H + (size_t)n * 64;
    float s1 = 0.0f, s2 = 0.0f;
    #pragma unroll 8
    for (int c = 0; c < 64; ++c) { const float h = hr[c]; s1 += h * a1[c]; s2 += h * a2[c]; }
    f1[n] = s1; f2[n] = s2;
}

// ---------------------------------------------------------------------------
// STGCN time block: relu(conv1 + sigmoid(conv2) + conv3), kernel (1,3) pad 1.
// Writes bf16 flattened (b,n,t*64+o) = GAT input (A operand, row-major).
// ---------------------------------------------------------------------------
__global__ void time_block(const float* __restrict__ x,
                           const float* __restrict__ w1, const float* __restrict__ b1,
                           const float* __restrict__ w2, const float* __restrict__ b2,
                           const float* __restrict__ w3, const float* __restrict__ b3,
                           __bf16* __restrict__ out, int T, int Cin, int total)
{
    const int idx = blockIdx.x * blockDim.x + threadIdx.x;
    if (idx >= total) return;
    const int o  = idx & 63;
    const int t  = (idx >> 6) % T;
    const int bn = idx / (64 * T);
    const float* xb = x + (size_t)bn * T * Cin;
    float a1 = 0.0f, a2 = 0.0f, a3 = 0.0f;
    for (int dt = 0; dt < 3; ++dt) {
        const int tt = t + dt - 1;
        if (tt < 0 || tt >= T) continue;
        const float* xr = xb + (size_t)tt * Cin;
        for (int c = 0; c < Cin; ++c) {
            const float  xv = xr[c];
            const size_t wi = ((size_t)o * Cin + c) * 3 + dt;
            a1 += xv * w1[wi];
            a2 += xv * w2[wi];
            a3 += xv * w3[wi];
        }
    }
    const float sig = 1.0f / (1.0f + __expf(-(a2 + b2[o])));
    const float r   = fmaxf(0.0f, (a1 + b1[o]) + sig + (a3 + b3[o]));
    out[idx] = (__bf16)r;
}

// ---------------------------------------------------------------------------
// Per-head epilogue: elu(hp + bias + res + db) -> concat slot or scaled add.
// ---------------------------------------------------------------------------
__global__ void head_epi(const float* __restrict__ Hp, const float* __restrict__ res,
                         const float* __restrict__ bias, const float* __restrict__ db,
                         float* __restrict__ t2, int Nn, int ch, int off, float scale,
                         int accum, int total)
{
    const int idx = blockIdx.x * blockDim.x + threadIdx.x;
    if (idx >= total) return;
    const int rowc = idx >> 6;
    const int c    = idx & 63;
    const int n    = rowc % Nn;
    float v = Hp[idx] + bias[n * 64 + c] + res[idx] + db[c];
    v = v > 0.0f ? v : (__expf(v) - 1.0f);
    const size_t o = (size_t)rowc * ch + off + c;
    if (accum) t2[o] += v * scale; else t2[o] = v;
}

// ---------------------------------------------------------------------------
// Residual add + BatchNorm2d(num_nodes) + relu. One block per node.
// ---------------------------------------------------------------------------
__global__ void __launch_bounds__(256)
combine_bn(const float* __restrict__ t2, const float* __restrict__ resid,
           const float* __restrict__ g, const float* __restrict__ bb,
           float* __restrict__ out, int Bb, int Nn, int Tc, int Tp)
{
    __shared__ float ssum[256], ssq[256];
    const int n   = blockIdx.x;
    const int ch  = Tc * 64;
    const int cnt = Bb * ch;
    float s = 0.0f, q = 0.0f;
    for (int idx = threadIdx.x; idx < cnt; idx += 256) {
        const int b = idx / ch, rem = idx - b * ch;
        float v = t2[(size_t)(b * Nn + n) * ch + rem];
        if (resid) {
            const int t = rem >> 6, c = rem & 63;
            v += resid[((size_t)(b * Nn + n) * Tp + (Tp - Tc + t)) * 64 + c];
        }
        s += v; q += v * v;
    }
    ssum[threadIdx.x] = s; ssq[threadIdx.x] = q;
    __syncthreads();
    for (int st = 128; st > 0; st >>= 1) {
        if (threadIdx.x < st) { ssum[threadIdx.x] += ssum[threadIdx.x + st];
                                ssq[threadIdx.x]  += ssq[threadIdx.x + st]; }
        __syncthreads();
    }
    const float mean = ssum[0] / (float)cnt;
    const float var  = ssq[0] / (float)cnt - mean * mean;
    const float rstd = rsqrtf(var + 1e-5f);
    const float gg = g[n], bv = bb[n];
    for (int idx = threadIdx.x; idx < cnt; idx += 256) {
        const int b = idx / ch, rem = idx - b * ch;
        float v = t2[(size_t)(b * Nn + n) * ch + rem];
        if (resid) {
            const int t = rem >> 6, c = rem & 63;
            v += resid[((size_t)(b * Nn + n) * Tp + (Tp - Tc + t)) * 64 + c];
        }
        v = (v - mean) * rstd * gg + bv;
        out[(size_t)(b * Nn + n) * ch + rem] = fmaxf(v, 0.0f);
    }
}

// ---------------------------------------------------------------------------
// GLU gate combine:  y = (out+cb)*sig(gate+gb) + res*(1-sig)
// ---------------------------------------------------------------------------
__global__ void glu_elem(const float* __restrict__ outA, const float* __restrict__ gateA,
                         const float* __restrict__ resA, const float* __restrict__ xin,
                         const float* __restrict__ cb, const float* __restrict__ gb,
                         const float* __restrict__ db, float* __restrict__ y,
                         int O, int total)
{
    const int idx = blockIdx.x * blockDim.x + threadIdx.x;
    if (idx >= total) return;
    const int o = idx % O;
    const float ov = outA[idx] + cb[o];
    const float gv = 1.0f / (1.0f + __expf(-(gateA[idx] + gb[o])));
    const float rv = resA ? (resA[idx] + db[o]) : xin[idx];
    y[idx] = ov * gv + rv * (1.0f - gv);
}

// ---------------------------------------------------------------------------
// BatchNorm1d over rows per channel + relu; writes fp32 and bf16 copies.
// ---------------------------------------------------------------------------
__global__ void __launch_bounds__(256)
bn_relu_ch(float* __restrict__ y, __bf16* __restrict__ ybf,
           const float* __restrict__ g, const float* __restrict__ bb, int M, int O)
{
    __shared__ float ssum[256], ssq[256];
    const int o = blockIdx.x;
    float s = 0.0f, q = 0.0f;
    for (int r = threadIdx.x; r < M; r += 256) {
        const float v = y[(size_t)r * O + o]; s += v; q += v * v;
    }
    ssum[threadIdx.x] = s; ssq[threadIdx.x] = q;
    __syncthreads();
    for (int st = 128; st > 0; st >>= 1) {
        if (threadIdx.x < st) { ssum[threadIdx.x] += ssum[threadIdx.x + st];
                                ssq[threadIdx.x]  += ssq[threadIdx.x + st]; }
        __syncthreads();
    }
    const float mean = ssum[0] / (float)M;
    const float var  = ssq[0] / (float)M - mean * mean;
    const float rstd = rsqrtf(var + 1e-5f);
    const float gg = g[o], bv = bb[o];
    for (int r = threadIdx.x; r < M; r += 256) {
        float v = (y[(size_t)r * O + o] - mean) * rstd * gg + bv;
        v = fmaxf(v, 0.0f);
        y[(size_t)r * O + o]   = v;
        ybf[(size_t)r * O + o] = (__bf16)v;
    }
}

// ---------------------------------------------------------------------------
// Utilities: A-side fp32->bf16; B-side pack into WMMA fragment order.
// ---------------------------------------------------------------------------
__global__ void f2bf(const float* __restrict__ src, __bf16* __restrict__ dst, int n)
{
    const int i = blockIdx.x * blockDim.x + threadIdx.x;
    if (i < n) dst[i] = (__bf16)src[i];
}

// row-major (K x N) fp32 -> packed bf16: dst[((k>>5)*N + c)*32 + (k&31)]
__global__ void pack_rowmajor_bf(const float* __restrict__ src, __bf16* __restrict__ dst,
                                 int K, int N)
{
    const int idx = blockIdx.x * blockDim.x + threadIdx.x;
    if (idx >= K * N) return;
    const int k = idx / N, c = idx - k * N;
    dst[((((size_t)(k >> 5)) * N + c) << 5) + (k & 31)] = (__bf16)src[idx];
}

// conv weight tap (O,I,ks)[:,:,ko], transposed to (I,O), packed:
// B[k=i][c=o] = src[(o*I+i)*ks+ko]
__global__ void pack_transpose_bf(const float* __restrict__ src, __bf16* __restrict__ dst,
                                  int O, int I, int ks, int ko)
{
    const int idx = blockIdx.x * blockDim.x + threadIdx.x;
    if (idx >= I * O) return;
    const int i = idx / O, o = idx - i * O;
    dst[((((size_t)(i >> 5)) * O + o) << 5) + (i & 31)] =
        (__bf16)src[((size_t)o * I + i) * ks + ko];
}

__global__ void fill_zero(float* __restrict__ p, int n)
{
    const int i = blockIdx.x * blockDim.x + threadIdx.x;
    if (i < n) p[i] = 0.0f;
}

__global__ void fill_zero_bf(__bf16* __restrict__ p, int n)
{
    const int i = blockIdx.x * blockDim.x + threadIdx.x;
    if (i < n) p[i] = (__bf16)0.0f;
}

// ---------------------------------------------------------------------------
// Host orchestration
// ---------------------------------------------------------------------------
extern "C" void kernel_launch(void* const* d_in, const int* in_sizes, int n_in,
                              void* d_out, int out_size, void* d_ws, size_t ws_size,
                              hipStream_t stream)
{
    (void)in_sizes; (void)n_in; (void)out_size; (void)ws_size;

    // ---- input walk (dict insertion order, depth-first) ----
    int ii = 0;
    const float* A_hat = (const float*)d_in[ii++];
    const float* X     = (const float*)d_in[ii++];

    const int nheads[4] = {4, 4, 4, 6};
    struct HeadP { const float *W, *a1, *a2, *bias, *dw, *db; };
    struct BlkP  { const float *w1, *b1, *w2, *b2, *w3, *b3; HeadP hd[6]; const float *g, *bb; };
    BlkP blk[4];
    for (int b = 0; b < 4; ++b) {
        blk[b].w1 = (const float*)d_in[ii++]; blk[b].b1 = (const float*)d_in[ii++];
        blk[b].w2 = (const float*)d_in[ii++]; blk[b].b2 = (const float*)d_in[ii++];
        blk[b].w3 = (const float*)d_in[ii++]; blk[b].b3 = (const float*)d_in[ii++];
        for (int h = 0; h < nheads[b]; ++h) {
            blk[b].hd[h].W    = (const float*)d_in[ii++];
            blk[b].hd[h].a1   = (const float*)d_in[ii++];
            blk[b].hd[h].a2   = (const float*)d_in[ii++];
            blk[b].hd[h].bias = (const float*)d_in[ii++];
            blk[b].hd[h].dw   = (const float*)d_in[ii++];
            blk[b].hd[h].db   = (const float*)d_in[ii++];
        }
        blk[b].g = (const float*)d_in[ii++]; blk[b].bb = (const float*)d_in[ii++];
    }
    struct GluP { const float *cw, *cb, *gw, *gb, *dw, *db, *g, *bb; };
    GluP glu[3];
    for (int q = 0; q < 3; ++q) {
        glu[q].cw = (const float*)d_in[ii++]; glu[q].cb = (const float*)d_in[ii++];
        glu[q].gw = (const float*)d_in[ii++]; glu[q].gb = (const float*)d_in[ii++];
        glu[q].dw = (const float*)d_in[ii++]; glu[q].db = (const float*)d_in[ii++];
        glu[q].g  = (const float*)d_in[ii++]; glu[q].bb = (const float*)d_in[ii++];
    }
    const float* ow = (const float*)d_in[ii++];
    const float* ob = (const float*)d_in[ii++];

    // ---- workspace carve-out ----
    const int Bb = 8, Nn = 1024, M = Bb * Nn;
    char* ws = (char*)d_ws;
    size_t off = 0;
    auto alloc = [&](size_t bytes) -> void* {
        void* p = ws + off;
        off = (off + bytes + 255) & ~(size_t)255;
        return p;
    };
    __bf16* zbuf   = (__bf16*)alloc((size_t)4096 * 2);      // zero operand scratch
    __bf16* tb_bf  = (__bf16*)alloc((size_t)M * 768 * 2);   // A: time-block out
    __bf16* wstage = (__bf16*)alloc((size_t)512 * 512 * 2); // packed B staging
    float*  h_f    = (float*) alloc((size_t)M * 64 * 4);
    __bf16* h_pk   = (__bf16*)alloc((size_t)M * 64 * 2);    // packed h (B operand)
    float*  f1     = (float*) alloc((size_t)M * 4);
    float*  f2     = (float*) alloc((size_t)M * 4);
    float*  cm     = (float*) alloc((size_t)M * 4);
    float*  rs     = (float*) alloc((size_t)M * 4);
    float*  hp     = (float*) alloc((size_t)M * 64 * 4);
    float*  resb   = (float*) alloc((size_t)M * 64 * 4);
    float*  t2     = (float*) alloc((size_t)M * 256 * 4);
    float*  xcur   = (float*) alloc((size_t)M * 4 * 64 * 4);
    float*  xnext  = (float*) alloc((size_t)M * 4 * 64 * 4);
    __bf16* embbf  = (__bf16*)alloc((size_t)M * 64 * 2);
    float*  accO   = (float*) alloc((size_t)M * 512 * 4);
    float*  accG   = (float*) alloc((size_t)M * 512 * 4);
    float*  accR   = (float*) alloc((size_t)M * 512 * 4);
    float*  y1     = (float*) alloc((size_t)M * 512 * 4);
    float*  y2     = (float*) alloc((size_t)M * 512 * 4);
    __bf16* ybfA   = (__bf16*)alloc((size_t)M * 512 * 2);
    __bf16* ybfB   = (__bf16*)alloc((size_t)M * 512 * 2);

    auto g1d = [](int n) { return dim3((n + 255) / 256); };

    fill_zero_bf<<<g1d(4096), 256, 0, stream>>>(zbuf, 4096);

    // ---- 4 STGAT blocks ----
    const int  TinArr[4]  = {12, 4, 4, 4};
    const int  CinArr[4]  = {2, 64, 64, 64};
    const bool concat[4]  = {true, true, true, false};
    const float* xin = X;
    int Tp = 12;
    for (int b = 0; b < 4; ++b) {
        const int Tin = TinArr[b], Cin = CinArr[b], gin = Tin * 64, nh = nheads[b];
        const int ch  = concat[b] ? nh * 64 : 64;
        const int Tc  = ch / 64;

        const int totTb = M * Tin * 64;
        time_block<<<g1d(totTb), 256, 0, stream>>>(
            xin, blk[b].w1, blk[b].b1, blk[b].w2, blk[b].b2, blk[b].w3, blk[b].b3,
            tb_bf, Tin, Cin, totTb);

        if (!concat[b])
            fill_zero<<<g1d(M * ch), 256, 0, stream>>>(t2, M * ch);

        const dim3 gH((M + 255) / 256, 1);
        for (int h = 0; h < nh; ++h) {
            const HeadP& H = blk[b].hd[h];
            // h = t @ W   (W is (gin,64) row-major -> pack)
            pack_rowmajor_bf<<<g1d(gin * 64), 256, 0, stream>>>(H.W, wstage, gin, 64);
            gemm_bf16_wmma<<<gH, 256, 0, stream>>>(tb_bf, wstage, zbuf, h_f, nullptr,
                                                   M, 64, gin, M, 0, 0);
            pack_rowmajor_bf<<<g1d(M * 64), 256, 0, stream>>>(h_f, h_pk, M, 64);
            // attention logits pieces + softmax column stats (axis=1)
            f12_kernel<<<g1d(M), 256, 0, stream>>>(h_f, H.a1, H.a2, f1, f2, M);
            col_stats<<<dim3(Nn / 256, Bb), 256, 0, stream>>>(A_hat, f1, f2, cm, rs, Nn);
            // fused softmax @ h
            att_gemm_wmma<<<dim3(Nn / 128, Bb), 256, 0, stream>>>(
                A_hat, f1, f2, cm, rs, h_pk, hp, Nn);
            // residual downsample: res = t @ dw^T
            pack_transpose_bf<<<g1d(gin * 64), 256, 0, stream>>>(H.dw, wstage, 64, gin, 1, 0);
            gemm_bf16_wmma<<<gH, 256, 0, stream>>>(tb_bf, wstage, zbuf, resb, nullptr,
                                                   M, 64, gin, M, 0, 0);
            head_epi<<<g1d(M * 64), 256, 0, stream>>>(
                hp, resb, H.bias, H.db, t2, Nn, ch,
                concat[b] ? h * 64 : 0, concat[b] ? 1.0f : (1.0f / nh),
                concat[b] ? 0 : 1, M * 64);
        }

        float* xout = (b % 2 == 0) ? xcur : xnext;
        combine_bn<<<Nn, 256, 0, stream>>>(t2, (b == 0) ? nullptr : xin,
                                           blk[b].g, blk[b].bb, xout, Bb, Nn, Tc, Tp);
        xin = xout;
        Tp = Tc;
    }

    // ---- end conv (GLU stack, layout (b*N+l, channel)) ----
    f2bf<<<g1d(M * 64), 256, 0, stream>>>(xin, embbf, M * 64);
    const dim3 gG((M + 255) / 256, 8);   // N = 512
    const dim3 gN((M + 255) / 256, 1);   // N <= 64

    // GLU1: k=1, 64 -> 512, with downsample residual
    pack_transpose_bf<<<g1d(512 * 64), 256, 0, stream>>>(glu[0].cw, wstage, 512, 64, 1, 0);
    gemm_bf16_wmma<<<gG, 256, 0, stream>>>(embbf, wstage, zbuf, accO, nullptr, M, 512, 64, M, 0, 0);
    pack_transpose_bf<<<g1d(512 * 64), 256, 0, stream>>>(glu[0].gw, wstage, 512, 64, 1, 0);
    gemm_bf16_wmma<<<gG, 256, 0, stream>>>(embbf, wstage, zbuf, accG, nullptr, M, 512, 64, M, 0, 0);
    pack_transpose_bf<<<g1d(512 * 64), 256, 0, stream>>>(glu[0].dw, wstage, 512, 64, 1, 0);
    gemm_bf16_wmma<<<gG, 256, 0, stream>>>(embbf, wstage, zbuf, accR, nullptr, M, 512, 64, M, 0, 0);
    glu_elem<<<g1d(M * 512), 256, 0, stream>>>(accO, accG, accR, nullptr,
                                               glu[0].cb, glu[0].gb, glu[0].db,
                                               y1, 512, M * 512);
    bn_relu_ch<<<512, 256, 0, stream>>>(y1, ybfA, glu[0].g, glu[0].bb, M, 512);

    // GLU2: k=3 causal, 512 -> 512 (3 row-shifted GEMMs per conv)
    for (int t = 0; t < 3; ++t) {
        pack_transpose_bf<<<g1d(512 * 512), 256, 0, stream>>>(glu[1].cw, wstage, 512, 512, 3, t);
        gemm_bf16_wmma<<<gG, 256, 0, stream>>>(ybfA, wstage, zbuf, accO, nullptr,
                                               M, 512, 512, Nn, t - 2, (t > 0));
    }
    for (int t = 0; t < 3; ++t) {
        pack_transpose_bf<<<g1d(512 * 512), 256, 0, stream>>>(glu[1].gw, wstage, 512, 512, 3, t);
        gemm_bf16_wmma<<<gG, 256, 0, stream>>>(ybfA, wstage, zbuf, accG, nullptr,
                                               M, 512, 512, Nn, t - 2, (t > 0));
    }
    glu_elem<<<g1d(M * 512), 256, 0, stream>>>(accO, accG, nullptr, y1,
                                               glu[1].cb, glu[1].gb, nullptr,
                                               y2, 512, M * 512);
    bn_relu_ch<<<512, 256, 0, stream>>>(y2, ybfB, glu[1].g, glu[1].bb, M, 512);

    // GLU3: k=3 causal, 512 -> 512
    for (int t = 0; t < 3; ++t) {
        pack_transpose_bf<<<g1d(512 * 512), 256, 0, stream>>>(glu[2].cw, wstage, 512, 512, 3, t);
        gemm_bf16_wmma<<<gG, 256, 0, stream>>>(ybfB, wstage, zbuf, accO, nullptr,
                                               M, 512, 512, Nn, t - 2, (t > 0));
    }
    for (int t = 0; t < 3; ++t) {
        pack_transpose_bf<<<g1d(512 * 512), 256, 0, stream>>>(glu[2].gw, wstage, 512, 512, 3, t);
        gemm_bf16_wmma<<<gG, 256, 0, stream>>>(ybfB, wstage, zbuf, accG, nullptr,
                                               M, 512, 512, Nn, t - 2, (t > 0));
    }
    glu_elem<<<g1d(M * 512), 256, 0, stream>>>(accO, accG, nullptr, y2,
                                               glu[2].cb, glu[2].gb, nullptr,
                                               y1, 512, M * 512);
    bn_relu_ch<<<512, 256, 0, stream>>>(y1, ybfA, glu[2].g, glu[2].bb, M, 512);

    // Final 1x1 conv 512 -> Tout(12); output layout (B,N,Tout) == d_out flat
    pack_transpose_bf<<<g1d(12 * 512), 256, 0, stream>>>(ow, wstage, 12, 512, 1, 0);
    gemm_bf16_wmma<<<gN, 256, 0, stream>>>(ybfA, wstage, zbuf, (float*)d_out, ob,
                                           M, 12, 512, M, 0, 0);
}
